// GrandDiffuser_20057497273094
// MI455X (gfx1250) — matrix operand order repaired and checked by reading, not compiled, not run on hardware
//
#include <hip/hip_runtime.h>
#include <hip/hip_bf16.h>

// Problem constants (match reference)
#define B_   8
#define C_   1024
#define DIN_ 1024
#define H_   16
#define DH_  64
#define OUT_ 1024
#define M_   (B_*C_)      // 8192 rows for the big GEMMs

typedef _Float16 v8h  __attribute__((ext_vector_type(8)));
typedef _Float16 v16h __attribute__((ext_vector_type(16)));
typedef float    v4f  __attribute__((ext_vector_type(4)));
typedef float    v8f  __attribute__((ext_vector_type(8)));

union F16x16 { v16h v; v8h h[2]; _Float16 e[16]; };

__device__ __forceinline__ v8f wmma16x16x32(v16h a, v16h b, v8f c) {
  // (neg_a, A, neg_b, B, c_mod, C, reuse_a, reuse_b)
  return __builtin_amdgcn_wmma_f32_16x16x32_f16(false, a, false, b, (short)0, c, false, false);
}

// ---------------------------------------------------------------------------
// Weight convert: W f32 [K][N] row-major -> Wt f16 [N][K] (transposed),
// done once so GEMM B-fragments become contiguous 16B global loads.
// ---------------------------------------------------------------------------
__global__ __launch_bounds__(1024) void wt_convert_kernel(
    const float* __restrict__ W, _Float16* __restrict__ Wt)
{
  __shared__ float tile[32][33];
  const int tx = threadIdx.x, ty = threadIdx.y;
  const int k0 = blockIdx.x * 32, n0 = blockIdx.y * 32;
  tile[ty][tx] = W[(size_t)(k0 + ty) * OUT_ + n0 + tx];
  __syncthreads();
  Wt[(size_t)(n0 + ty) * DIN_ + k0 + tx] = (_Float16)tile[tx][ty];
}

// ---------------------------------------------------------------------------
// Barrier-free direct-fragment GEMM. Block = 8 waves covering 128x128;
// each wave owns a 32x64 tile (2x4 WMMA fragments, 8 WMMA per K-step).
// All fragment loads are contiguous global_load_b128s (L0/L2 resident:
// weights = 2MB f16, activations stream once; MI455X L2 = 192MB).
// ---------------------------------------------------------------------------

// QKV projection: A f32 [M][K] (converted to f16 in-register),
// Bt f16 [N][K], bias f32 -> f16 stored [b][h][c][d] (+ optional [b][h][d][c]).
template <bool WRITE_T>
__global__ __launch_bounds__(256) void gemm_qkv_direct(
    const float* __restrict__ A, const _Float16* __restrict__ Bt,
    const float* __restrict__ bias, _Float16* __restrict__ outStd,
    _Float16* __restrict__ outT)
{
  const int tid  = threadIdx.x;
  const int lane = tid & 31;
  const int wave = tid >> 5;
  const int l16  = lane & 15;
  const int lhi  = lane >> 4;
  const int m_blk = blockIdx.x * 128 + (wave >> 1) * 32;   // wave's 32 rows
  const int n_blk = blockIdx.y * 128 + (wave & 1) * 64;    // wave's 64 cols
  const int aoff = lhi * 8;    // ISA A layout: lanes>=16 start at K=8
  const int koff = lhi * 8;    // ISA B layout: lanes>=16 start at K=8

  v8f acc[2][4];
  for (int i = 0; i < 2; ++i)
    for (int j = 0; j < 4; ++j) {
      float bval = bias[n_blk + j*16 + l16];
      for (int r = 0; r < 8; ++r) acc[i][j][r] = bval;
    }

  const float* arow[2] = {
    A + (size_t)(m_blk +      l16) * DIN_,
    A + (size_t)(m_blk + 16 + l16) * DIN_ };
  const _Float16* brow[4];
  for (int j = 0; j < 4; ++j)
    brow[j] = Bt + (size_t)(n_blk + j*16 + l16) * DIN_;

  for (int k0 = 0; k0 < DIN_; k0 += 32) {
    F16x16 a[2];
    for (int i = 0; i < 2; ++i) {
      const v4f* ap = (const v4f*)(arow[i] + k0 + aoff);
      v4f x0 = ap[0], x1 = ap[1];          // K = aoff..aoff+7
      v4f x2 = ap[4], x3 = ap[5];          // K = 16+aoff..16+aoff+7
      for (int j = 0; j < 4; ++j) {
        a[i].e[j]      = (_Float16)x0[j];
        a[i].e[4 + j]  = (_Float16)x1[j];
        a[i].e[8 + j]  = (_Float16)x2[j];
        a[i].e[12 + j] = (_Float16)x3[j];
      }
    }
    F16x16 b[4];
    for (int j = 0; j < 4; ++j) {
      b[j].h[0] = *(const v8h*)&brow[j][k0 + koff];
      b[j].h[1] = *(const v8h*)&brow[j][k0 + koff + 16];
    }
    for (int i = 0; i < 2; ++i)
      for (int j = 0; j < 4; ++j)
        acc[i][j] = wmma16x16x32(a[i].v, b[j].v, acc[i][j]);
  }

  for (int i = 0; i < 2; ++i)
    for (int j = 0; j < 4; ++j)
      for (int r = 0; r < 8; ++r) {
        int rg = m_blk + i*16 + r + lhi*8;   // global row (b*C + c)
        int cg = n_blk + j*16 + l16;         // global col (h*64 + d)
        int bb = rg >> 10, cc = rg & 1023;
        int hh = cg >> 6,  dd = cg & 63;
        _Float16 val = (_Float16)acc[i][j][r];
        outStd[(((size_t)bb*H_ + hh)*C_ + cc)*DH_ + dd] = val;
        if constexpr (WRITE_T)
          outT[(((size_t)bb*H_ + hh)*DH_ + dd)*C_ + cc] = val;
      }
}

// Output projection: A f16 [M][K] @ Wt f16 [N][K] + bias -> f32 out [M][N].
__global__ __launch_bounds__(256) void gemm_out_direct(
    const _Float16* __restrict__ A, const _Float16* __restrict__ Bt,
    const float* __restrict__ bias, float* __restrict__ out)
{
  const int tid  = threadIdx.x;
  const int lane = tid & 31;
  const int wave = tid >> 5;
  const int l16  = lane & 15;
  const int lhi  = lane >> 4;
  const int m_blk = blockIdx.x * 128 + (wave >> 1) * 32;
  const int n_blk = blockIdx.y * 128 + (wave & 1) * 64;
  const int aoff = lhi * 8;
  const int koff = lhi * 8;

  v8f acc[2][4];
  for (int i = 0; i < 2; ++i)
    for (int j = 0; j < 4; ++j) {
      float bval = bias[n_blk + j*16 + l16];
      for (int r = 0; r < 8; ++r) acc[i][j][r] = bval;
    }

  const _Float16* arow[2] = {
    A + (size_t)(m_blk +      l16) * DIN_,
    A + (size_t)(m_blk + 16 + l16) * DIN_ };
  const _Float16* brow[4];
  for (int j = 0; j < 4; ++j)
    brow[j] = Bt + (size_t)(n_blk + j*16 + l16) * DIN_;

  for (int k0 = 0; k0 < DIN_; k0 += 32) {
    F16x16 a[2];
    for (int i = 0; i < 2; ++i) {
      a[i].h[0] = *(const v8h*)&arow[i][k0 + aoff];
      a[i].h[1] = *(const v8h*)&arow[i][k0 + 16 + aoff];
    }
    F16x16 b[4];
    for (int j = 0; j < 4; ++j) {
      b[j].h[0] = *(const v8h*)&brow[j][k0 + koff];
      b[j].h[1] = *(const v8h*)&brow[j][k0 + koff + 16];
    }
    for (int i = 0; i < 2; ++i)
      for (int j = 0; j < 4; ++j)
        acc[i][j] = wmma16x16x32(a[i].v, b[j].v, acc[i][j]);
  }

  for (int i = 0; i < 2; ++i)
    for (int j = 0; j < 4; ++j)
      for (int r = 0; r < 8; ++r) {
        int rg = m_blk + i*16 + r + lhi*8;
        int cg = n_blk + j*16 + l16;
        out[(size_t)rg*OUT_ + cg] = acc[i][j][r];
      }
}

// ---------------------------------------------------------------------------
// Attention: one block per (b, h, 64-row c tile). Full score strip S[64][1024]
// f32 in LDS (~265KB of the WGP's 320KB). Exact softmax, then P@V - V.
// ---------------------------------------------------------------------------
#define SROW 1032                     // padded row stride (floats)
#define SCR  (64 * SROW)              // reduction scratch offset

__global__ __launch_bounds__(256) void attn_kernel(
    const _Float16* __restrict__ khf, const _Float16* __restrict__ qhf,
    const _Float16* __restrict__ vhf, const _Float16* __restrict__ vhT,
    _Float16* __restrict__ Y)
{
  __shared__ __align__(16) float S[64 * SROW + 256];
  const int tid  = threadIdx.x;
  const int lane = tid & 31;
  const int wave = tid >> 5;
  const int l16  = lane & 15;
  const int lhi  = lane >> 4;
  const int c0 = blockIdx.x * 64;
  const int h  = blockIdx.y;
  const int b  = blockIdx.z;
  const size_t bh = (size_t)b * H_ + h;
  const _Float16* Kp = khf + bh * C_ * DH_;   // [c][d]
  const _Float16* Qp = qhf + bh * C_ * DH_;   // [q][d]
  const _Float16* Vp = vhf + bh * C_ * DH_;   // [q][d]
  const _Float16* Vt = vhT + bh * DH_ * C_;   // [d][q]

  // ---- Phase 1: S[c][q] = sum_d K[c][d] * Q[q][d] ----
  {
    const int tm = wave & 3;                    // this wave's 16-row strip
    F16x16 aK[2];
    const int arow = c0 + tm*16 + l16;
    const int aoff = lhi * 8;
    for (int kk = 0; kk < 2; ++kk) {            // K-dim (d) = 64 -> two chunks
      aK[kk].h[0] = *(const v8h*)&Kp[(size_t)arow*DH_ + kk*32 + aoff];
      aK[kk].h[1] = *(const v8h*)&Kp[(size_t)arow*DH_ + kk*32 + 16 + aoff];
    }
    const int koff = lhi * 8;
    for (int t = wave; t < 256; t += 8) {       // 32 q-tiles per wave
      const int q0   = (t >> 2) * 16;
      const int qcol = q0 + l16;
      F16x16 bQ;
      v8f s = {};
      bQ.h[0] = *(const v8h*)&Qp[(size_t)qcol*DH_ + koff];
      bQ.h[1] = *(const v8h*)&Qp[(size_t)qcol*DH_ + koff + 16];
      s = wmma16x16x32(aK[0].v, bQ.v, s);
      bQ.h[0] = *(const v8h*)&Qp[(size_t)qcol*DH_ + 32 + koff];
      bQ.h[1] = *(const v8h*)&Qp[(size_t)qcol*DH_ + 32 + koff + 16];
      s = wmma16x16x32(aK[1].v, bQ.v, s);
      for (int r = 0; r < 8; ++r)
        S[(tm*16 + r + lhi*8)*SROW + q0 + l16] = s[r];
    }
  }
  __syncthreads();

  // ---- Phase 2: exact softmax along q (1024); 4 threads per row ----
  {
    const int row = tid >> 2;
    const int seg = tid & 3;
    float* rp = &S[row*SROW + seg*256];
    float m = -3.0e38f;
    for (int j = 0; j < 256; ++j) m = fmaxf(m, rp[j]);
    S[SCR + tid] = m;
    __syncthreads();
    const int rb = SCR + (row << 2);
    float rm = fmaxf(fmaxf(S[rb], S[rb+1]), fmaxf(S[rb+2], S[rb+3]));
    __syncthreads();
    float sum = 0.f;
    for (int j = 0; j < 256; ++j) { float e = __expf(rp[j] - rm); rp[j] = e; sum += e; }
    S[SCR + tid] = sum;
    __syncthreads();
    float rs  = S[rb] + S[rb+1] + S[rb+2] + S[rb+3];
    float inv = 1.0f / rs;
    for (int j = 0; j < 256; ++j) rp[j] *= inv;
  }
  __syncthreads();

  // ---- Phase 3: O[c][d] = sum_q P[c][q] V[q][d] - V[c][d] ----
  {
    const int t0  = wave * 2;           // two 16x16 tiles per wave (64x64 out)
    const int m0  = (t0 >> 2) * 16;
    const int n0a = (t0 & 3) * 16;
    const int n0b = n0a + 16;
    const int arow = m0 + l16;
    const int aoff = lhi * 8;
    const int koff = lhi * 8;
    const _Float16* vtA = Vt + (size_t)(n0a + l16) * C_;
    const _Float16* vtB = Vt + (size_t)(n0b + l16) * C_;
    v8f accA = {}, accB = {};
    for (int kq = 0; kq < C_; kq += 32) {
      F16x16 a;
      const float* sp = &S[arow*SROW + kq + aoff];
      for (int j = 0; j < 8; ++j) a.e[j]     = (_Float16)sp[j];
      for (int j = 0; j < 8; ++j) a.e[8 + j] = (_Float16)sp[16 + j];
      F16x16 bA, bB;
      bA.h[0] = *(const v8h*)&vtA[kq + koff];
      bA.h[1] = *(const v8h*)&vtA[kq + koff + 16];
      bB.h[0] = *(const v8h*)&vtB[kq + koff];
      bB.h[1] = *(const v8h*)&vtB[kq + koff + 16];
      accA = wmma16x16x32(a.v, bA.v, accA);
      accB = wmma16x16x32(a.v, bB.v, accB);
    }
    for (int r = 0; r < 8; ++r) {
      int crow = m0 + r + lhi*8;
      int cg = c0 + crow;
      int d0 = n0a + l16;
      int d1 = n0b + l16;
      float o0 = accA[r] - (float)Vp[(size_t)cg*DH_ + d0];
      float o1 = accB[r] - (float)Vp[(size_t)cg*DH_ + d1];
      size_t yrow = (size_t)(b*C_ + cg) * (size_t)(H_*DH_);
      Y[yrow + h*DH_ + d0] = (_Float16)o0;
      Y[yrow + h*DH_ + d1] = (_Float16)o1;
    }
  }
}

// ---------------------------------------------------------------------------
// Host launch
// ---------------------------------------------------------------------------
extern "C" void kernel_launch(void* const* d_in, const int* in_sizes, int n_in,
                              void* d_out, int out_size, void* d_ws, size_t ws_size,
                              hipStream_t stream) {
  (void)in_sizes; (void)n_in; (void)out_size; (void)ws_size;
  // setup_inputs order: t(0, unused), k(1), q(2), v(3), Wk(4), bk(5),
  //                     Wq(6), bq(7), Wv(8), bv(9), Wo(10), bo(11)
  const float* kin = (const float*)d_in[1];
  const float* qin = (const float*)d_in[2];
  const float* vin = (const float*)d_in[3];
  const float* Wk  = (const float*)d_in[4];
  const float* bk  = (const float*)d_in[5];
  const float* Wq  = (const float*)d_in[6];
  const float* bq  = (const float*)d_in[7];
  const float* Wv  = (const float*)d_in[8];
  const float* bv  = (const float*)d_in[9];
  const float* Wo  = (const float*)d_in[10];
  const float* bo  = (const float*)d_in[11];
  float* out = (float*)d_out;

  // Workspace layout (f16): 5 x 8M activation buffers + 4 x 1M weights = 88MB.
  const size_t elems = (size_t)B_ * H_ * C_ * DH_;   // 8,388,608
  const size_t welems = (size_t)DIN_ * OUT_;         // 1,048,576
  _Float16* khf = (_Float16*)d_ws;
  _Float16* qhf = khf + elems;
  _Float16* vhf = qhf + elems;
  _Float16* vhT = vhf + elems;   // transposed V: [b][h][d][c]
  _Float16* Yb  = vhT + elems;   // attention output, [b*C][H*DH]
  _Float16* WkT = Yb  + elems;   // f16 transposed weights [n][k]
  _Float16* WqT = WkT + welems;
  _Float16* WvT = WqT + welems;
  _Float16* WoT = WvT + welems;

  // 1) Convert + transpose all weights to f16 [N][K].
  dim3 tblk(32, 32);
  dim3 tgrid(DIN_/32, OUT_/32);
  wt_convert_kernel<<<tgrid, tblk, 0, stream>>>(Wk, WkT);
  wt_convert_kernel<<<tgrid, tblk, 0, stream>>>(Wq, WqT);
  wt_convert_kernel<<<tgrid, tblk, 0, stream>>>(Wv, WvT);
  wt_convert_kernel<<<tgrid, tblk, 0, stream>>>(Wo, WoT);

  // 2) QKV projections (barrier-free direct-fragment WMMA GEMMs).
  dim3 blk(256);
  dim3 gG(M_/128, OUT_/128);     // 64 x 8
  gemm_qkv_direct<false><<<gG, blk, 0, stream>>>(kin, WkT, bk, khf, (_Float16*)nullptr);
  gemm_qkv_direct<false><<<gG, blk, 0, stream>>>(qin, WqT, bq, qhf, (_Float16*)nullptr);
  gemm_qkv_direct<true ><<<gG, blk, 0, stream>>>(vin, WvT, bv, vhf, vhT);

  // 3) Attention (scores + exact softmax + P@V - V) per (b,h,c-tile).
  dim3 gA(C_/64, H_, B_);        // 16 x 16 x 8
  attn_kernel<<<gA, blk, 0, stream>>>(khf, qhf, vhf, vhT, Yb);

  // 4) Output projection -> f32 d_out.
  gemm_out_direct<<<gG, blk, 0, stream>>>(Yb, WoT, bo, out);
}
